// FlashRetentionBody_82240033783988
// MI455X (gfx1250) — compile-verified, matrix operand order checked by reading, not run on hardware
//
#include <hip/hip_runtime.h>

// FlashRetentionBody for MI455X (gfx1250, wave32).
// Pass 1: per-16-row stripe, row abs-sums via v_wmma_f32_16x16x32_f16 with B=ones
//         (f32 accumulation chained through C), cross-wave reduce in LDS.
// Pass 2: streaming normalize; stripe re-read of qk/m hits L2 (192 MB), so HBM
//         traffic stays at the ~640 MiB roofline minimum (~29 us at 23.3 TB/s).

typedef _Float16 half_t;
typedef __attribute__((ext_vector_type(16))) _Float16      v16h;
typedef __attribute__((ext_vector_type(8)))  _Float16      v8h;
typedef __attribute__((ext_vector_type(8)))  float         v8f;
typedef __attribute__((ext_vector_type(4)))  unsigned int  v4u;

#define N_COLS         8192
#define ROWS_PER_BLOCK 16
#define THREADS        256
#define WAVES          8
#define COLS_PER_WAVE  (N_COLS / WAVES)   // 1024

union ATile  { v16h h; v4u u[2]; unsigned int w[8]; };  // 16 halves / lane = WMMA A operand
union HTile8 { v8h  h; v4u u;    unsigned int w[4]; };  // 8 halves (one 128-bit vmem op)

__global__ __launch_bounds__(THREADS)
void flash_retention_kernel(const half_t* __restrict__ qk,
                            const half_t* __restrict__ m,
                            const half_t* __restrict__ acco,
                            const half_t* __restrict__ rvec,
                            const half_t* __restrict__ r_wo_clamp,
                            half_t* __restrict__ out_acco,
                            half_t* __restrict__ out_acc,
                            half_t* __restrict__ out_rnew)
{
    __shared__ float partials[WAVES][ROWS_PER_BLOCK];
    __shared__ float s_acc [ROWS_PER_BLOCK];   // 1/r_new
    __shared__ float s_acco[ROWS_PER_BLOCK];   // r/r_new

    const int tid  = threadIdx.x;
    const int lane = tid & 31;
    const int wave = tid >> 5;
    const int r0   = blockIdx.x * ROWS_PER_BLOCK;

    // ------------- Pass 1: row abs-sums with WMMA (A = |qk*m|, B = ones) -------------
    // 16-bit A layout: lane L holds only elements of matrix row L%16;
    // lanes 0-15 hold K = {0..7, 16..23}, lanes 16-31 hold K = {8..15, 24..31}.
    const int    rowL    = lane & 15;
    const size_t rowBase = (size_t)(r0 + rowL) * N_COLS;
    const int    koff    = (lane >> 4) << 3;   // 0 or 8 halves

    v16h bones;
#pragma unroll
    for (int i = 0; i < 16; ++i) bones[i] = (_Float16)1.0f;

    v8f c = {};                                // f32 accumulator (16x16)
    const int c0 = wave * COLS_PER_WAVE;
    for (int cc = 0; cc < COLS_PER_WAVE; cc += 32) {
        const size_t a0 = rowBase + (size_t)(c0 + cc) + (size_t)koff;
        ATile q, mm, p;
        q.u[0]  = *(const v4u*)(qk + a0);        // global_load_b128
        q.u[1]  = *(const v4u*)(qk + a0 + 16);
        mm.u[0] = *(const v4u*)(m  + a0);
        mm.u[1] = *(const v4u*)(m  + a0 + 16);
        p.h = q.h * mm.h;                        // v_pk_mul_f16
#pragma unroll
        for (int i = 0; i < 8; ++i) p.w[i] &= 0x7FFF7FFFu;   // packed |.| (|a*b|=|a||b|)
        c = __builtin_amdgcn_wmma_f32_16x16x32_f16(false, p.h, false, bones,
                                                   (short)0, c, false, false);
    }

    // C/D layout: lanes 0-15 -> rows 0-7 (VGPR k = row k); lanes 16-31 -> rows 8-15.
    if (lane == 0) {
#pragma unroll
        for (int k = 0; k < 8; ++k) partials[wave][k] = c[k];
    } else if (lane == 16) {
#pragma unroll
        for (int k = 0; k < 8; ++k) partials[wave][8 + k] = c[k];
    }
    __syncthreads();

    if (tid < ROWS_PER_BLOCK) {
        float s = 0.f;
#pragma unroll
        for (int w = 0; w < WAVES; ++w) s += partials[w][tid];
        const int   gr   = r0 + tid;
        const float rnew = fmaxf((float)r_wo_clamp[gr] + s, 1.0f);
        out_rnew[gr] = (half_t)rnew;
        s_acc [tid] = 1.0f / rnew;
        s_acco[tid] = (float)rvec[gr] / rnew;
    }
    __syncthreads();

    // ------------- Pass 2: streaming normalize (qk/m stripe re-read hits L2) -------------
    const size_t stripeBase = (size_t)r0 * N_COLS;
#pragma unroll 4
    for (int it = 0; it < (ROWS_PER_BLOCK * N_COLS) / (THREADS * 8); ++it) {  // 64
        const int    idx = (it * THREADS + tid) * 8;
        const int    rl  = idx >> 13;            // / 8192 (groups never straddle rows)
        const size_t g   = stripeBase + (size_t)idx;
        HTile8 q, mm, a, pacc, pacco;
        q.u  = *(const v4u*)(qk   + g);
        mm.u = *(const v4u*)(m    + g);
        a.u  = *(const v4u*)(acco + g);
        const _Float16 sa = (_Float16)s_acc [rl];
        const _Float16 so = (_Float16)s_acco[rl];
        v8h sav, sov;
#pragma unroll
        for (int i = 0; i < 8; ++i) { sav[i] = sa; sov[i] = so; }
        pacc.h  = q.h * mm.h * sav;              // acc      = qkm / r_new
        pacco.h = a.h * sov;                     // acco_out = acco * r / r_new
        *(v4u*)(out_acc  + g) = pacc.u;          // global_store_b128
        *(v4u*)(out_acco + g) = pacco.u;
    }
}

extern "C" void kernel_launch(void* const* d_in, const int* in_sizes, int n_in,
                              void* d_out, int out_size, void* d_ws, size_t ws_size,
                              hipStream_t stream) {
    (void)in_sizes; (void)n_in; (void)out_size; (void)d_ws; (void)ws_size;
    const half_t* qk   = (const half_t*)d_in[0];
    const half_t* m    = (const half_t*)d_in[1];
    const half_t* acco = (const half_t*)d_in[2];
    const half_t* r    = (const half_t*)d_in[3];
    const half_t* rwo  = (const half_t*)d_in[4];

    half_t* out = (half_t*)d_out;
    const size_t MN = (size_t)8192 * 8192;
    half_t* out_acco = out;            // 64M fp16
    half_t* out_acc  = out + MN;       // 64M fp16
    half_t* out_rnew = out + 2 * MN;   // 8192 fp16

    dim3 grid(8192 / ROWS_PER_BLOCK);  // 512 blocks, 8 waves each
    dim3 block(THREADS);
    flash_retention_kernel<<<grid, block, 0, stream>>>(qk, m, acco, r, rwo,
                                                       out_acco, out_acc, out_rnew);
}
